// DependencyParser_63574105916159
// MI455X (gfx1250) — matrix operand order repaired and checked by reading, not compiled, not run on hardware
//
#include <hip/hip_runtime.h>
#include <math.h>

typedef __attribute__((ext_vector_type(16))) __bf16 v16bf;
typedef __attribute__((ext_vector_type(8)))  __bf16 v8bf;
typedef __attribute__((ext_vector_type(8)))  float  v8f;

union VU16 { v16bf v; v8bf h[2]; };

__device__ __forceinline__ float sigf(float x){ return 1.0f/(1.0f + __expf(-x)); }

// ---------------------------------------------------------------------------
// whhT[k][j] = whh[j][k]   (whh: 1024x256 -> whhT: 256x1024), one-time 1MB
// ---------------------------------------------------------------------------
__global__ void k_transpose_whh(const float* __restrict__ whh, float* __restrict__ whhT){
    int o = blockIdx.x * blockDim.x + threadIdx.x;   // 262144 elements
    int k = o >> 10, j = o & 1023;
    whhT[o] = whh[j * 256 + k];
}

__global__ void k_zero(float* __restrict__ p, int n){
    int i = blockIdx.x * blockDim.x + threadIdx.x;
    if (i < n) p[i] = 0.f;
}

// ---------------------------------------------------------------------------
// One-time: w2 (128x256 f32) -> fragment-ready bf16. For B-frag (32x16 tile
// nt, k-tile kt) lane's 16 elements are w2[n][kt*32 + hi*16 + 0..15] with
// n = nt*16 + (lane&15): k-contiguous, so store them contiguous per lane.
// Layout: w2s[(((nt*8)+kt)*32 + lane)*16 + e]
// ---------------------------------------------------------------------------
__global__ void k_swz_w2(const float* __restrict__ w2, __bf16* __restrict__ w2s){
    int idx = blockIdx.x * blockDim.x + threadIdx.x;   // 32768
    int e    = idx & 15;
    int lane = (idx >> 4) & 31;
    int kt   = (idx >> 9) & 7;
    int nt   = (idx >> 12) & 7;
    int n = nt * 16 + (lane & 15);
    int k = kt * 32 + (lane >> 4) * 16 + e;
    w2s[idx] = (__bf16)w2[n * 256 + k];
}

// ---------------------------------------------------------------------------
// Generic GEMM: C[M x N] = A[M x K] @ W[N x K]^T (+bias1+bias2), bf16 WMMA.
// Grid (M/32, N/64), block 256 (8 waves). k-loop split: full tiles use
// unguarded float4 staging + float2 B loads; one guarded remainder tile.
// ---------------------------------------------------------------------------
__global__ void k_gemm_bf16(const float* __restrict__ A, int K,
                            const float* __restrict__ W, int ldw,
                            const float* __restrict__ bias1, const float* __restrict__ bias2,
                            float* __restrict__ C, int ldc){
    __shared__ float As[32][36];                 // padded: bank-conflict-free column gather
    const int tid  = threadIdx.x;
    const int wv   = tid >> 5, lane = tid & 31;
    const int l15  = lane & 15;
    const int hi   = (lane >> 4) & 1;
    const int mtile = wv & 1;
    const int ntile = wv >> 1;
    const int rowBase = blockIdx.x * 32;
    const int n = blockIdx.y * 64 + ntile * 16 + l15;
    const int m = mtile * 16 + l15;
    const int srow = tid >> 3, scol = (tid & 7) * 4;   // float4 staging coords

    v8f acc = {0.f,0.f,0.f,0.f,0.f,0.f,0.f,0.f};
    const int kfull = K >> 5;
    const int rem   = K & 31;

    for (int kt = 0; kt < kfull; ++kt){
        // unguarded staging: one b128 global load + one b128 LDS store per thread
        *(float4*)&As[srow][scol] =
            *(const float4*)(A + (rowBase + srow) * K + kt * 32 + scol);
        __syncthreads();
        v16bf af;
        #pragma unroll
        for (int ii = 0; ii < 8; ++ii){
            int kb = ((ii < 4) ? 0 : 16) + (hi ? 8 : 0) + (ii & 3) * 2;
            af[2*ii]   = (__bf16)As[m][kb];
            af[2*ii+1] = (__bf16)As[m][kb+1];
        }
        v16bf bf;
        #pragma unroll
        for (int v = 0; v < 8; ++v){
            int kk = kt * 32 + ((v < 4) ? 0 : 16) + (hi ? 8 : 0) + (v & 3) * 2;
            float2 bv = *(const float2*)(W + n * ldw + kk);
            bf[2*v]   = (__bf16)bv.x;
            bf[2*v+1] = (__bf16)bv.y;
        }
        acc = __builtin_amdgcn_wmma_f32_16x16x32_bf16(false, af, false, bf,
                                                      (short)0, acc, false, false);
        __syncthreads();
    }
    if (rem){                                    // K=200 tail: kc < 8 valid
        for (int idx = tid; idx < 1024; idx += 256){
            int r = idx >> 5, kc = idx & 31;
            As[r][kc] = (kc < rem) ? A[(rowBase + r) * K + kfull * 32 + kc] : 0.f;
        }
        __syncthreads();
        v16bf af;
        #pragma unroll
        for (int ii = 0; ii < 8; ++ii){
            int kb = ((ii < 4) ? 0 : 16) + (hi ? 8 : 0) + (ii & 3) * 2;
            af[2*ii]   = (__bf16)As[m][kb];
            af[2*ii+1] = (__bf16)As[m][kb+1];
        }
        v16bf bf;
        #pragma unroll
        for (int v = 0; v < 8; ++v){
            int kc = ((v < 4) ? 0 : 16) + (hi ? 8 : 0) + (v & 3) * 2;
            float2 bv = make_float2(0.f, 0.f);
            if (kc + 1 < rem) bv = *(const float2*)(W + n * ldw + kfull * 32 + kc);
            bf[2*v]   = (__bf16)bv.x;
            bf[2*v+1] = (__bf16)bv.y;
        }
        acc = __builtin_amdgcn_wmma_f32_16x16x32_bf16(false, af, false, bf,
                                                      (short)0, acc, false, false);
    }
    float bia = 0.f;
    if (bias1) bia += bias1[n];
    if (bias2) bia += bias2[n];
    #pragma unroll
    for (int r = 0; r < 8; ++r){                 // D: m = r + 8*hi, n = lane&15
        int row = rowBase + mtile * 16 + r + (hi ? 8 : 0);
        C[row * ldc + n] = acc[r] + bia;
    }
}

// ---------------------------------------------------------------------------
// LSTM scan, both directions in ONE launch (grid.x = 2).
// ---------------------------------------------------------------------------
__global__ void __launch_bounds__(1024) k_lstm_pair(
        const float* __restrict__ xw0, const float* __restrict__ xw1,
        const float* __restrict__ wT0, const float* __restrict__ wT1,
        float* __restrict__ out /* 768 x 512 */){
    const int dir = blockIdx.x;
    const float* xw = dir ? xw1 : xw0;
    const float* wT = dir ? wT1 : wT0;
    const int colOff = dir ? 256 : 0;
    __shared__ float h[256], c[256], g[1024];
    const int j = threadIdx.x;
    if (j < 256){ h[j] = 0.f; c[j] = 0.f; }
    __syncthreads();
    for (int s = 0; s < 768; ++s){
        int t = dir ? (767 - s) : s;
        float acc = xw[t * 1024 + j];
        #pragma unroll 8
        for (int k = 0; k < 256; ++k)
            acc = fmaf(h[k], wT[k * 1024 + j], acc);
        g[j] = acc;
        __syncthreads();
        if (j < 256){
            float gi = sigf(g[j]);
            float gf = sigf(g[256 + j]);
            float gg = tanhf(g[512 + j]);
            float go = sigf(g[768 + j]);
            float cn = gf * c[j] + gi * gg;
            float hn = go * tanhf(cn);
            c[j] = cn; h[j] = hn;
            out[t * 512 + colOff + j] = hn;
        }
        __syncthreads();
    }
}

// ---------------------------------------------------------------------------
// Biaffine scorer: block = (i, 32 j-rows). H1 built straight to bf16 in LDS
// (A-frag spans are contiguous 8-element runs; row stride 264 bf16 keeps
// 16B alignment and staggers banks). w2 comes pre-swizzled: one 32B load per
// fragment. Inner loop: 2 ds_load_b128 + 4 global_load_b128 + 2 wmma per kt.
// ---------------------------------------------------------------------------
__global__ void k_scorer(const float* __restrict__ aM, const float* __restrict__ cM,
                         const float* __restrict__ b1, const __bf16* __restrict__ w2s,
                         const float* __restrict__ b2, const float* __restrict__ w3,
                         const float* __restrict__ b3, float* __restrict__ score,
                         float* __restrict__ denom){
    __shared__ __bf16 H1b[32 * 264];
    __shared__ float rawAcc[32];
    const int tid = threadIdx.x;
    const int i  = blockIdx.x;
    const int jb = blockIdx.y * 32;
    if (tid < 32) rawAcc[tid] = 0.f;
    {
        float at  = aM[i * 256 + tid];
        float b1t = b1[tid];
        #pragma unroll 4
        for (int e = 0; e < 32; ++e){
            float v = at + cM[(jb + e) * 256 + tid] + b1t;
            H1b[e * 264 + tid] = (__bf16)(v > 0.f ? v : 0.f);
        }
    }
    __syncthreads();

    const int wv = tid >> 5, lane = tid & 31;
    const int l15 = lane & 15, hi = (lane >> 4) & 1;
    const int mtile = wv & 1;                    // 2 m-tiles
    const int nt0 = (wv >> 1) * 2;               // 8 n-tiles, 2 per wave
    const int m  = mtile * 16 + l15;
    const int n0 = nt0 * 16 + l15, n1 = n0 + 16;
    const __bf16* rowA = &H1b[m * 264];
    const __bf16* w2f0 = w2s + (( nt0      * 8) * 32 + lane) * 16;
    const __bf16* w2f1 = w2s + (((nt0 + 1) * 8) * 32 + lane) * 16;

    v8f acc0 = {0.f,0.f,0.f,0.f,0.f,0.f,0.f,0.f};
    v8f acc1 = {0.f,0.f,0.f,0.f,0.f,0.f,0.f,0.f};
    #pragma unroll
    for (int kt = 0; kt < 8; ++kt){              // K = 256
        VU16 a;
        a.h[0] = *(const v8bf*)(rowA + kt * 32 + hi * 16);      // elems 0..7
        a.h[1] = *(const v8bf*)(rowA + kt * 32 + hi * 16 + 8);  // elems 8..15
        v16bf bf0 = *(const v16bf*)(w2f0 + kt * 512);
        v16bf bf1 = *(const v16bf*)(w2f1 + kt * 512);
        acc0 = __builtin_amdgcn_wmma_f32_16x16x32_bf16(false, a.v, false, bf0,
                                                       (short)0, acc0, false, false);
        acc1 = __builtin_amdgcn_wmma_f32_16x16x32_bf16(false, a.v, false, bf1,
                                                       (short)0, acc1, false, false);
    }
    // h2 = relu(acc + b2); contract with w3; butterfly-reduce over the 16 n's
    float w3a = w3[n0], b2a = b2[n0];
    float w3b = w3[n1], b2b = b2[n1];
    #pragma unroll
    for (int r = 0; r < 8; ++r){
        float h2a = acc0[r] + b2a; h2a = h2a > 0.f ? h2a : 0.f;
        float h2b = acc1[r] + b2b; h2b = h2b > 0.f ? h2b : 0.f;
        float p = h2a * w3a + h2b * w3b;
        p += __shfl_xor(p, 1, 32);
        p += __shfl_xor(p, 2, 32);
        p += __shfl_xor(p, 4, 32);
        p += __shfl_xor(p, 8, 32);
        if (l15 == 0) atomicAdd(&rawAcc[mtile * 16 + r + hi * 8], p);
    }
    __syncthreads();
    if (tid < 32){
        int jg = jb + tid;
        float raw = rawAcc[tid] + b3[0];
        raw = raw > 0.f ? raw : 0.f;
        bool valid = (jg >= 1) && (jg != i);
        float sc = valid ? raw : 0.f;
        score[i * 768 + jg] = sc;
        if (jg != i) atomicAdd(&denom[jg], expf(sc));   // denom[j] = sum_{i!=j} e^score
    }
}

// ---------------------------------------------------------------------------
// loss = mean_t( log(denom[dep_t]) - score[head_t, dep_t] ), t = 1..767
// ---------------------------------------------------------------------------
__global__ void k_loss(const int* __restrict__ tree, const float* __restrict__ denom,
                       const float* __restrict__ score, float* __restrict__ out0){
    __shared__ float red[256];
    int tid = threadIdx.x;
    float s = 0.f;
    for (int t = 1 + tid; t < 768; t += 256){
        int v1 = tree[2 * t];
        int v2 = tree[2 * t + 1];
        s += logf(denom[v2]) - score[v1 * 768 + v2];
    }
    red[tid] = s;
    __syncthreads();
    for (int off = 128; off > 0; off >>= 1){
        if (tid < off) red[tid] += red[tid + off];
        __syncthreads();
    }
    if (tid == 0) out0[0] = red[0] / 767.f;
}

// ---------------------------------------------------------------------------
extern "C" void kernel_launch(void* const* d_in, const int* in_sizes, int n_in,
                              void* d_out, int out_size, void* d_ws, size_t ws_size,
                              hipStream_t stream){
    (void)in_sizes; (void)n_in; (void)out_size; (void)ws_size;
    const float* x       = (const float*)d_in[0];
    const int*   tree    = (const int*)  d_in[1];
    const float* wih_l0f = (const float*)d_in[2];
    const float* whh_l0f = (const float*)d_in[3];
    const float* bih_l0f = (const float*)d_in[4];
    const float* bhh_l0f = (const float*)d_in[5];
    const float* wih_l0b = (const float*)d_in[6];
    const float* whh_l0b = (const float*)d_in[7];
    const float* bih_l0b = (const float*)d_in[8];
    const float* bhh_l0b = (const float*)d_in[9];
    const float* wih_l1f = (const float*)d_in[10];
    const float* whh_l1f = (const float*)d_in[11];
    const float* bih_l1f = (const float*)d_in[12];
    const float* bhh_l1f = (const float*)d_in[13];
    const float* wih_l1b = (const float*)d_in[14];
    const float* whh_l1b = (const float*)d_in[15];
    const float* bih_l1b = (const float*)d_in[16];
    const float* bhh_l1b = (const float*)d_in[17];
    const float* w1 = (const float*)d_in[18];
    const float* b1 = (const float*)d_in[19];
    const float* w2 = (const float*)d_in[20];
    const float* b2 = (const float*)d_in[21];
    const float* w3 = (const float*)d_in[22];
    const float* b3 = (const float*)d_in[23];

    float* ws   = (float*)d_ws;
    float* xwf  = ws;                       // 768*1024
    float* xwb  = xwf  + 768 * 1024;        // 768*1024
    float* h0   = xwb  + 768 * 1024;        // 768*512
    float* emb  = h0   + 768 * 512;         // 768*512
    float* aM   = emb  + 768 * 512;         // 768*256
    float* cM   = aM   + 768 * 256;         // 768*256
    float* wT0f = cM   + 768 * 256;         // 4 x 262144
    float* wT0b = wT0f + 262144;
    float* wT1f = wT0b + 262144;
    float* wT1b = wT1f + 262144;
    float* denom = wT1b + 262144;           // 768
    __bf16* w2s  = (__bf16*)(denom + 768);  // 32768 bf16 (32B-aligned offset)

    float* out   = (float*)d_out;
    float* score = out + 1;                 // d_out = [loss, score(768x768)]

    // one-time prep (independent)
    k_transpose_whh<<<1024, 256, 0, stream>>>(whh_l0f, wT0f);
    k_transpose_whh<<<1024, 256, 0, stream>>>(whh_l0b, wT0b);
    k_transpose_whh<<<1024, 256, 0, stream>>>(whh_l1f, wT1f);
    k_transpose_whh<<<1024, 256, 0, stream>>>(whh_l1b, wT1b);
    k_zero<<<3, 256, 0, stream>>>(denom, 768);
    k_swz_w2<<<128, 256, 0, stream>>>(w2, w2s);

    // layer 0: XW GEMMs (WMMA) then serial scan (both dirs in one launch)
    dim3 gXW(24, 16);                       // 768x1024
    k_gemm_bf16<<<gXW, 256, 0, stream>>>(x, 200, wih_l0f, 200, bih_l0f, bhh_l0f, xwf, 1024);
    k_gemm_bf16<<<gXW, 256, 0, stream>>>(x, 200, wih_l0b, 200, bih_l0b, bhh_l0b, xwb, 1024);
    k_lstm_pair<<<2, 1024, 0, stream>>>(xwf, xwb, wT0f, wT0b, h0);

    // layer 1
    k_gemm_bf16<<<gXW, 256, 0, stream>>>(h0, 512, wih_l1f, 512, bih_l1f, bhh_l1f, xwf, 1024);
    k_gemm_bf16<<<gXW, 256, 0, stream>>>(h0, 512, wih_l1b, 512, bih_l1b, bhh_l1b, xwb, 1024);
    k_lstm_pair<<<2, 1024, 0, stream>>>(xwf, xwb, wT1f, wT1b, emb);

    // a = emb @ w1[:, :512].T ; c = emb @ w1[:, 512:].T  (bias b1 applied in scorer)
    dim3 gAC(24, 4);                        // 768x256
    k_gemm_bf16<<<gAC, 256, 0, stream>>>(emb, 512, w1,       1024, nullptr, nullptr, aM, 256);
    k_gemm_bf16<<<gAC, 256, 0, stream>>>(emb, 512, w1 + 512, 1024, nullptr, nullptr, cM, 256);

    // pairwise MLP scorer (dominant compute, bf16 WMMA) + loss
    k_scorer<<<dim3(768, 24), 256, 0, stream>>>(aM, cM, b1, w2s, b2, w3, b3, score, denom);
    k_loss<<<1, 256, 0, stream>>>(tree, denom, score, out);
}
